// NaiveAttention_14869176779279
// MI455X (gfx1250) — matrix operand order repaired and checked by reading, not compiled, not run on hardware
//
#include <hip/hip_runtime.h>

// ---------------------------------------------------------------------------
// MI455X (gfx1250) fused attention, f16 WMMA (V_WMMA_F32_16X16X32_F16).
//   convert x->f16, W->f16 transposed (one-shot)
//   3x GEMM (async double-buffered LDS staging, wave32 WMMA)  -> q, k, vT(f16)
//   flash attention (512 thr / 16 waves): scores never hit HBM, K/V frags
//     straight from L2, 16x1024 f32 accumulator in registers across 16 waves,
//     wave-per-row softmax with shfl_xor butterflies
//   GEMM -> f32 output projection
// ---------------------------------------------------------------------------

typedef _Float16 h8   __attribute__((ext_vector_type(8)));
typedef _Float16 v16h __attribute__((ext_vector_type(16)));
typedef float    v8f  __attribute__((ext_vector_type(8)));

#define AS3 __attribute__((address_space(3)))

#if __has_builtin(__builtin_amdgcn_sched_barrier)
#define SCHED_FENCE() __builtin_amdgcn_sched_barrier(0)
#else
#define SCHED_FENCE()
#endif

union Frag {
  v16h v;
  h8   h[2];
};

__device__ __forceinline__ v8f wmma_f16f32(v16h a, v16h b, v8f c) {
  // (neg_a, A, neg_b, B, c_mod, C, reuse_a, reuse_b)
  return __builtin_amdgcn_wmma_f32_16x16x32_f16(false, a, false, b, (short)0, c,
                                                false, false);
}

// A-matrix fragment (16x32 f16), row-major source, row stride `stride` elems.
// row = lane&15; lanes<16 hold K=0..15, lanes>=16 hold K=16..31 (contiguous).
__device__ __forceinline__ v16h load_afrag(const _Float16* p0, int stride, int lane) {
  const _Float16* p = p0 + (size_t)(lane & 15) * stride + ((lane >> 4) * 16);
  Frag f;
  f.h[0] = *(const h8*)p;
  f.h[1] = *(const h8*)(p + 8);
  return f.v;
}

// B-matrix fragment (32x16 f16) from B^T storage [N][K] (row stride `stride`).
// col = lane&15; lanes<16 hold K=0..15, lanes>=16 hold K=16..31 (contiguous).
__device__ __forceinline__ v16h load_btfrag(const _Float16* p0, int stride, int lane) {
  const _Float16* p = p0 + (size_t)(lane & 15) * stride + ((lane >> 4) * 16);
  Frag f;
  f.h[0] = *(const h8*)p;
  f.h[1] = *(const h8*)(p + 8);
  return f.v;
}

// CDNA5 async copy: 16B global -> LDS, tracked by ASYNCcnt (no VGPR data path).
__device__ __forceinline__ void async_copy16(_Float16* lds, const _Float16* g) {
  unsigned loff = (unsigned)(size_t)(AS3 _Float16*)lds;  // wave-relative LDS addr
  asm volatile("global_load_async_to_lds_b128 %0, %1, off"
               :: "v"(loff), "v"(g) : "memory");
}
__device__ __forceinline__ void wait_async(bool keep4) {
  if (keep4) asm volatile("s_wait_asynccnt 0x4" ::: "memory");  // next tile in flight
  else       asm volatile("s_wait_asynccnt 0x0" ::: "memory");
}

// ---------------------------------------------------------------------------
// One-shot converters.
// ---------------------------------------------------------------------------
__global__ void __launch_bounds__(256)
cvt_f16_kernel(const float* __restrict__ in, _Float16* __restrict__ out) {
  const size_t i = ((size_t)blockIdx.x * 256 + threadIdx.x) * 8;
  const float4* s = (const float4*)(in + i);
  float4 a = s[0], b = s[1];
  h8 o;
  o[0] = (_Float16)a.x; o[1] = (_Float16)a.y; o[2] = (_Float16)a.z; o[3] = (_Float16)a.w;
  o[4] = (_Float16)b.x; o[5] = (_Float16)b.y; o[6] = (_Float16)b.z; o[7] = (_Float16)b.w;
  *(h8*)(out + i) = o;
}

// W f32 [K=1024][N=1024] -> Wt f16 [N][K] (coalesced 32x32 LDS transpose)
__global__ void __launch_bounds__(256)
transpose_f16_kernel(const float* __restrict__ W, _Float16* __restrict__ Wt) {
  __shared__ _Float16 t[32][33];
  const int bx = blockIdx.x * 32, by = blockIdx.y * 32;
  const int tx = threadIdx.x & 31, ty = threadIdx.x >> 5;
#pragma unroll
  for (int rr = 0; rr < 32; rr += 8)
    t[ty + rr][tx] = (_Float16)W[(size_t)(by + ty + rr) * 1024 + bx + tx];
  __syncthreads();
#pragma unroll
  for (int rr = 0; rr < 32; rr += 8)
    Wt[(size_t)(bx + ty + rr) * 1024 + by + tx] = t[tx][ty + rr];
}

// ---------------------------------------------------------------------------
// GEMM: Out[M,N] = A[M,K] @ Bt^T + bias.  A f16 [M][K], Bt f16 [N][K].
// K = N = 1024, M = 16384. 256 thr / 8 waves, 128x128 tile, wave 64x32.
// Double-buffered LDS, async global->LDS staging, K-step 32.
// OUT_MODE: 0 = f32 row-major, 1 = f16 row-major, 2 = f16 transposed-per-batch.
// ---------------------------------------------------------------------------
template <int OUT_MODE>
__global__ void __launch_bounds__(256)
gemm_bias_kernel(const _Float16* __restrict__ A, const _Float16* __restrict__ Bt,
                 const float* __restrict__ bias, void* __restrict__ Out) {
  constexpr int K = 1024, N = 1024;
  constexpr int LS = 40;  // padded stride, rows stay 16B aligned
  __shared__ _Float16 As[2][128 * LS];
  __shared__ _Float16 Bs[2][128 * LS];

  const int tid  = threadIdx.x;
  const int lane = tid & 31;
  const int w    = tid >> 5;
  const int wm0  = (w >> 2) * 64;
  const int wn0  = (w & 3) * 32;
  const int m0   = blockIdx.x * 128;
  const int n0   = blockIdx.y * 128;
  const int c16  = lane & 15;

  // per-thread staging addresses: thread copies one half-row (16 f16 = 2x16B)
  const int sr = tid >> 1, sh = (tid & 1) * 16;
  const _Float16* ga0 = A  + (size_t)(m0 + sr) * K + sh;
  const _Float16* gb0 = Bt + (size_t)(n0 + sr) * K + sh;

  auto stage = [&](int buf, int k0) {
    _Float16* la = &As[buf][sr * LS + sh];
    _Float16* lb = &Bs[buf][sr * LS + sh];
    async_copy16(la,     ga0 + k0);
    async_copy16(la + 8, ga0 + k0 + 8);
    async_copy16(lb,     gb0 + k0);
    async_copy16(lb + 8, gb0 + k0 + 8);
  };

  v8f acc[4][2] = {};

  stage(0, 0);
  constexpr int KT = K / 32;
  for (int kt = 0; kt < KT; ++kt) {
    const int buf = kt & 1;
    if (kt + 1 < KT) stage(buf ^ 1, (kt + 1) * 32);
    wait_async(kt + 1 < KT);   // this buffer landed; next still streaming
    __syncthreads();

    // issue all fragment loads first, then the WMMA burst
    v16h bf[2], af[4];
#pragma unroll
    for (int ni = 0; ni < 2; ++ni)
      bf[ni] = load_btfrag(&Bs[buf][(wn0 + 16 * ni) * LS], LS, lane);
#pragma unroll
    for (int mi = 0; mi < 4; ++mi)
      af[mi] = load_afrag(&As[buf][(wm0 + 16 * mi) * LS], LS, lane);
    SCHED_FENCE();  // keep all loads ahead of the WMMA burst
#pragma unroll
    for (int mi = 0; mi < 4; ++mi)
#pragma unroll
      for (int ni = 0; ni < 2; ++ni)
        acc[mi][ni] = wmma_f16f32(af[mi], bf[ni], acc[mi][ni]);
    __syncthreads();  // LDS reuse guard (this buffer is restaged at kt+2)
  }

  // epilogue: +bias; C/D layout: VGPR r -> row (lane>>4)*8 + r, col lane&15
#pragma unroll
  for (int mi = 0; mi < 4; ++mi) {
#pragma unroll
    for (int ni = 0; ni < 2; ++ni) {
      const int col   = n0 + wn0 + 16 * ni + c16;
      const int row0g = m0 + wm0 + 16 * mi + (lane >> 4) * 8;
      const float bv  = bias[col];
#pragma unroll
      for (int r = 0; r < 8; ++r) {
        const float val = acc[mi][ni][r] + bv;
        const int row = row0g + r;
        if constexpr (OUT_MODE == 0) {
          ((float*)Out)[(size_t)row * N + col] = val;
        } else if constexpr (OUT_MODE == 1) {
          ((_Float16*)Out)[(size_t)row * N + col] = (_Float16)val;
        } else {  // per-batch transposed: vT[b][chan][seq]
          ((_Float16*)Out)[((size_t)(row >> 12) * 1024 + col) * 4096 + (row & 4095)] =
              (_Float16)val;
        }
      }
    }
  }
}

// ---------------------------------------------------------------------------
// Flash attention. Block = (batch, 16-query tile), 512 thr / 16 waves.
// Wave w owns channel slice [w*64, w*64+64): O (16x1024 f32) lives in
// registers across the 16 waves (4 x v8f per wave).
// K-frags from kh (row-major), V-frags from vT (chan-major): contiguous
// 32B/lane global loads served from L2. LDS only for score reduce + softmax.
// Softmax: one wave per query row, lane-parallel with shfl_xor butterflies.
// ---------------------------------------------------------------------------
__global__ void __launch_bounds__(512)
flash_attn_kernel(const _Float16* __restrict__ qh, const _Float16* __restrict__ kh,
                  const _Float16* __restrict__ vT, _Float16* __restrict__ oh) {
  constexpr int C = 1024, NSEQ = 4096, NW = 16;
  __shared__ float    Spart[NW * 512];  // 32 KB: per-wave partial scores (16x32)
  __shared__ float    S[512];
  __shared__ _Float16 Ph[512];
  __shared__ float    mrow[16], lrow[16], arow[16];

  const int tid  = threadIdx.x;
  const int lane = tid & 31;
  const int w    = tid >> 5;           // 0..15
  const int b    = blockIdx.y;
  const int m0   = blockIdx.x * 16;
  const size_t base  = (size_t)b * NSEQ * C;
  const size_t vbase = (size_t)b * C * NSEQ;
  const float scale  = 0.03125f;       // 1/sqrt(1024)
  const int row0  = (lane >> 4) * 8;
  const int c16   = lane & 15;
  const int cbase = w * 64;            // this wave's channel slice

  // resident Q fragments: 2 k-chunks of 32 channels
  v16h qf[2];
#pragma unroll
  for (int j = 0; j < 2; ++j)
    qf[j] = load_afrag(qh + base + (size_t)m0 * C + cbase + 32 * j, C, lane);

  v8f of[4] = {};
  if (tid < 16) { mrow[tid] = -1e30f; lrow[tid] = 0.f; }
  __syncthreads();

  for (int n0 = 0; n0 < NSEQ; n0 += 32) {
    // ---- scores: issue all 4 K-fragment loads, then 4 WMMAs
    v16h kf[2][2];  // [t][j]
#pragma unroll
    for (int t = 0; t < 2; ++t)
#pragma unroll
      for (int j = 0; j < 2; ++j)
        kf[t][j] = load_btfrag(
            kh + base + (size_t)(n0 + 16 * t) * C + cbase + 32 * j, C, lane);
    SCHED_FENCE();  // keep the 8 b128 loads ahead of the WMMAs
    v8f s[2] = {};
#pragma unroll
    for (int t = 0; t < 2; ++t)
#pragma unroll
      for (int j = 0; j < 2; ++j)
        s[t] = wmma_f16f32(qf[j], kf[t][j], s[t]);
#pragma unroll
    for (int t = 0; t < 2; ++t) {
      const int col = 16 * t + c16;
#pragma unroll
      for (int r = 0; r < 8; ++r) Spart[w * 512 + (row0 + r) * 32 + col] = s[t][r];
    }
    __syncthreads();
    {  // deterministic 16-wave reduction, 1 element/thread
      float acc = 0.f;
#pragma unroll
      for (int ww = 0; ww < NW; ++ww) acc += Spart[ww * 512 + tid];
      S[tid] = acc;
    }
    __syncthreads();
    // ---- online softmax: wave w owns query row w, one lane per key column
    {
      const float v    = S[w * 32 + lane] * scale;
      const float mold = mrow[w];
      float mx = v;
#pragma unroll
      for (int off = 16; off > 0; off >>= 1) mx = fmaxf(mx, __shfl_xor(mx, off, 32));
      mx = fmaxf(mx, mold);
      const float p = __expf(v - mx);
      float sum = p;
#pragma unroll
      for (int off = 16; off > 0; off >>= 1) sum += __shfl_xor(sum, off, 32);
      Ph[w * 32 + lane] = (_Float16)p;
      if (lane == 0) {
        const float alpha = __expf(mold - mx);
        mrow[w] = mx;
        lrow[w] = lrow[w] * alpha + sum;
        arow[w] = alpha;
      }
    }
    __syncthreads();
    // ---- rescale O, then O += P @ V (4 V-frag loads, then 4 WMMAs)
    {
      float a[8];
#pragma unroll
      for (int r = 0; r < 8; ++r) a[r] = arow[row0 + r];
#pragma unroll
      for (int i = 0; i < 4; ++i)
#pragma unroll
        for (int r = 0; r < 8; ++r) of[i][r] *= a[r];
    }
    const v16h pa = load_afrag(Ph, 32, lane);  // P tile 16x32
    v16h vf[4];
#pragma unroll
    for (int i = 0; i < 4; ++i)
      vf[i] = load_btfrag(vT + vbase + (size_t)(cbase + 16 * i) * NSEQ + n0,
                          NSEQ, lane);
    SCHED_FENCE();  // keep the 8 b128 loads ahead of the WMMAs
#pragma unroll
    for (int i = 0; i < 4; ++i) of[i] = wmma_f16f32(pa, vf[i], of[i]);
  }

  // ---- finalize: O /= l, store f16
  float linv[8];
#pragma unroll
  for (int r = 0; r < 8; ++r) linv[r] = 1.f / lrow[row0 + r];
#pragma unroll
  for (int i = 0; i < 4; ++i) {
    const int chan = cbase + 16 * i + c16;
#pragma unroll
    for (int r = 0; r < 8; ++r)
      oh[base + (size_t)(m0 + row0 + r) * C + chan] =
          (_Float16)(of[i][r] * linv[r]);
  }
}

// ---------------------------------------------------------------------------
extern "C" void kernel_launch(void* const* d_in, const int* in_sizes, int n_in,
                              void* d_out, int out_size, void* d_ws, size_t ws_size,
                              hipStream_t stream) {
  const float* x  = (const float*)d_in[0];
  const float* Wq = (const float*)d_in[1];
  const float* bq = (const float*)d_in[2];
  const float* Wk = (const float*)d_in[3];
  const float* bk = (const float*)d_in[4];
  const float* Wv = (const float*)d_in[5];
  const float* bv = (const float*)d_in[6];
  const float* Wo = (const float*)d_in[7];
  const float* bo = (const float*)d_in[8];

  const size_t MT = (size_t)16384 * 1024;  // B*N x C elements
  _Float16* xh  = (_Float16*)d_ws;
  _Float16* qh  = xh + MT;
  _Float16* kh  = qh + MT;
  _Float16* vT  = kh + MT;        // [b][chan][seq]
  _Float16* oh  = vT + MT;
  _Float16* Wqt = oh + MT;        // 4x 1Kx1K f16 transposed weights
  _Float16* Wkt = Wqt + (size_t)1024 * 1024;
  _Float16* Wvt = Wkt + (size_t)1024 * 1024;
  _Float16* Wot = Wvt + (size_t)1024 * 1024;

  const dim3 bb(256);
  cvt_f16_kernel<<<8192, bb, 0, stream>>>(x, xh);
  const dim3 tg(32, 32);
  transpose_f16_kernel<<<tg, bb, 0, stream>>>(Wq, Wqt);
  transpose_f16_kernel<<<tg, bb, 0, stream>>>(Wk, Wkt);
  transpose_f16_kernel<<<tg, bb, 0, stream>>>(Wv, Wvt);
  transpose_f16_kernel<<<tg, bb, 0, stream>>>(Wo, Wot);

  const dim3 gg(128, 8);
  gemm_bias_kernel<1><<<gg, bb, 0, stream>>>(xh, Wqt, bq, qh);
  gemm_bias_kernel<1><<<gg, bb, 0, stream>>>(xh, Wkt, bk, kh);
  gemm_bias_kernel<2><<<gg, bb, 0, stream>>>(xh, Wvt, bv, vT);
  flash_attn_kernel<<<dim3(256, 4), dim3(512), 0, stream>>>(qh, kh, vT, oh);
  gemm_bias_kernel<0><<<gg, bb, 0, stream>>>(oh, Wot, bo, d_out);

  (void)in_sizes; (void)n_in; (void)out_size; (void)ws_size;
}
